// HopfieldBuffer_3607772529107
// MI455X (gfx1250) — matrix operand order repaired and checked by reading, not compiled, not run on hardware
//
#include <hip/hip_runtime.h>
#include <hip/hip_bf16.h>

typedef __attribute__((ext_vector_type(16))) __bf16 v16bf;
typedef __attribute__((ext_vector_type(8)))  float  v8f;

#define N_BUF     100000
#define NQ        32
#define IMG_D     3072          // flattened image
#define D_FULL    3074          // + label + task
#define D_PAD     3104          // padded to 97 * 32
#define NCHUNK    97            // D_PAD / 32  (K chunks for QK^T)
#define KEY_TILES 6250          // N_BUF / 16
#define KCHUNKS   3125          // N_BUF / 32  (K chunks for PV)

// ---- workspace layout (bytes), total ~19.4 MB ----
#define WS_SCORES   0u          // 32 * 100000 f32 = 12,800,000
#define WS_ROWMAX   12800000u   // 32 f32
#define WS_ROWINV   12800128u   // 32 f32
#define WS_ROWVAL   12800256u   // 32 f32
#define WS_QOPS     12800512u   // 97*2*32*16 bf16 = 198,656
#define WS_WOPS     13000192u   // 3125*2*32*16 bf16 = 6,400,000

// A-operand (16-bit, 16x32) per-lane k mapping (contiguous, matches B):
// lanes 0-15 hold M=0-15 with K={0..15};
// lanes 16-31 same M with K={16..31}.
__device__ __forceinline__ int kmapA(int h, int lane) {
  return ((lane >= 16) ? 16 : 0) + h;
}

__device__ __forceinline__ v8f wmma_bf16(v16bf a, v16bf b, v8f c) {
  // D(16x16,f32) = A(16x32,bf16) x B(32x16,bf16) + C
  return __builtin_amdgcn_wmma_f32_16x16x32_bf16(false, a, false, b, (short)0, c,
                                                 false, false);
}

// -------- kernel 1: pack bf16 query into WMMA A-operand layout --------
__global__ void __launch_bounds__(256) build_q_ops(
    const float* __restrict__ X, const int* __restrict__ y,
    const int* __restrict__ task_ids, __bf16* __restrict__ q_ops) {
  int idx = blockIdx.x * 256 + threadIdx.x;
  if (idx >= NCHUNK * 2 * 32 * 16) return;
  int h    = idx & 15;
  int lane = (idx >> 4) & 31;
  int t    = (idx >> 9) & 1;           // query tile (q 0-15 / 16-31)
  int c    = idx >> 10;                // k-chunk
  int q    = t * 16 + (lane & 15);
  int k    = c * 32 + kmapA(h, lane);
  float v = 0.0f;
  if (k < IMG_D)            v = X[q * IMG_D + k];
  else if (k == IMG_D)      v = (float)y[q];
  else if (k == IMG_D + 1)  v = (float)task_ids[q];
  q_ops[idx] = (__bf16)v;
}

// -------- kernel 2: scores = scale * (Q . stored^T), masked --------
// one wave per 16-key tile: A = query tiles, B = 16 keys x 32 d-values
__global__ void __launch_bounds__(256) scores_kernel(
    const float* __restrict__ buf_X, const int* __restrict__ buf_y,
    const int* __restrict__ buf_task, const int* __restrict__ task_ids,
    const __bf16* __restrict__ q_ops, float* __restrict__ scores) {
  const int lane = threadIdx.x & 31;
  const int wave = threadIdx.x >> 5;
  const int tile = blockIdx.x * 8 + wave;
  if (tile >= KEY_TILES) return;                 // wave-uniform; EXEC stays full
  const int keyBase  = tile * 16;
  const int myKey    = keyBase + (lane & 15);    // B-operand: lane%16 = key col
  const int koffHalf = (lane >> 4) << 4;         // lanes 16-31 carry K+16

  v8f acc0 = {}, acc1 = {};
  const float* rowp = buf_X + (size_t)myKey * IMG_D + koffHalf;

  for (int c = 0; c < 96; ++c) {
    const v16bf a0 = *(const v16bf*)(q_ops + ((size_t)(c * 2 + 0) * 32 + lane) * 16);
    const v16bf a1 = *(const v16bf*)(q_ops + ((size_t)(c * 2 + 1) * 32 + lane) * 16);
    __builtin_prefetch((const void*)(rowp + (c + 4) * 32), 0, 1);
    const float4 f0 = *(const float4*)(rowp + c * 32 + 0);
    const float4 f1 = *(const float4*)(rowp + c * 32 + 4);
    const float4 f2 = *(const float4*)(rowp + c * 32 + 8);
    const float4 f3 = *(const float4*)(rowp + c * 32 + 12);
    v16bf b;
    b[0]  = (__bf16)f0.x; b[1]  = (__bf16)f0.y; b[2]  = (__bf16)f0.z; b[3]  = (__bf16)f0.w;
    b[4]  = (__bf16)f1.x; b[5]  = (__bf16)f1.y; b[6]  = (__bf16)f1.z; b[7]  = (__bf16)f1.w;
    b[8]  = (__bf16)f2.x; b[9]  = (__bf16)f2.y; b[10] = (__bf16)f2.z; b[11] = (__bf16)f2.w;
    b[12] = (__bf16)f3.x; b[13] = (__bf16)f3.y; b[14] = (__bf16)f3.z; b[15] = (__bf16)f3.w;
    acc0 = wmma_bf16(a0, b, acc0);
    acc1 = wmma_bf16(a1, b, acc1);
  }
  { // chunk 96: k = 3072 (label), 3073 (task), rest zero padding
    const v16bf a0 = *(const v16bf*)(q_ops + ((size_t)(96 * 2 + 0) * 32 + lane) * 16);
    const v16bf a1 = *(const v16bf*)(q_ops + ((size_t)(96 * 2 + 1) * 32 + lane) * 16);
    v16bf b;
#pragma unroll
    for (int j = 0; j < 16; ++j) b[j] = (__bf16)0.0f;
    if (lane < 16) {                       // K 0..15 of this chunk -> k 3072..3087
      b[0] = (__bf16)(float)buf_y[myKey];
      b[1] = (__bf16)(float)buf_task[myKey];
    }                                      // reconverged: EXEC full at WMMA
    acc0 = wmma_bf16(a0, b, acc0);
    acc1 = wmma_bf16(a1, b, acc1);
  }

  const float scale  = rsqrtf((float)D_FULL);
  const int   myTask = buf_task[myKey];
  const int   qAdd   = (lane >> 4) << 3;   // D layout: lanes 16-31 hold M+8
#pragma unroll
  for (int r = 0; r < 8; ++r) {
    const int q0 = r + qAdd, q1 = q0 + 16;
    float s0 = acc0[r] * scale;
    float s1 = acc1[r] * scale;
    if (task_ids[q0] == myTask) s0 = -1e30f;
    if (task_ids[q1] == myTask) s1 = -1e30f;
    scores[(size_t)q0 * N_BUF + myKey] = s0;
    scores[(size_t)q1 * N_BUF + myKey] = s1;
  }
}

// -------- kernel 3: per-query max / 1/sum(exp) / valid --------
__global__ void __launch_bounds__(256) softmax_stats(
    const float* __restrict__ scores, float* __restrict__ row_max,
    float* __restrict__ row_inv, float* __restrict__ row_valid) {
  __shared__ float sdata[256];
  const int q = blockIdx.x;
  const float* row = scores + (size_t)q * N_BUF;
  float m = -3e30f;
  for (int i = threadIdx.x; i < N_BUF; i += 256) m = fmaxf(m, row[i]);
  sdata[threadIdx.x] = m;
  __syncthreads();
  for (int s = 128; s > 0; s >>= 1) {
    if (threadIdx.x < s) sdata[threadIdx.x] = fmaxf(sdata[threadIdx.x], sdata[threadIdx.x + s]);
    __syncthreads();
  }
  const float mx = sdata[0];
  __syncthreads();
  float sum = 0.0f;
  for (int i = threadIdx.x; i < N_BUF; i += 256) sum += __expf(row[i] - mx);
  sdata[threadIdx.x] = sum;
  __syncthreads();
  for (int s = 128; s > 0; s >>= 1) {
    if (threadIdx.x < s) sdata[threadIdx.x] += sdata[threadIdx.x + s];
    __syncthreads();
  }
  if (threadIdx.x == 0) {
    row_max[q]   = mx;
    row_inv[q]   = 1.0f / sdata[0];
    row_valid[q] = (mx > -1e29f) ? 1.0f : 0.0f;
  }
}

// -------- kernel 4: softmax weights as bf16, pre-packed in A-operand layout ----
__global__ void __launch_bounds__(256) build_w_ops(
    const float* __restrict__ scores, const float* __restrict__ row_max,
    const float* __restrict__ row_inv, __bf16* __restrict__ w_ops) {
  int idx = blockIdx.x * 256 + threadIdx.x;
  if (idx >= KCHUNKS * 2 * 32 * 16) return;
  int h    = idx & 15;
  int lane = (idx >> 4) & 31;
  int t    = (idx >> 9) & 1;
  int c    = idx >> 10;
  int q    = t * 16 + (lane & 15);
  int key  = c * 32 + kmapA(h, lane);    // < 100000 always
  float w = __expf(scores[(size_t)q * N_BUF + key] - row_max[q]) * row_inv[q];
  w_ops[idx] = (__bf16)w;
}

// -------- kernel 5: sampled = W . stored  (one block per 32 output cols) ------
__global__ void __launch_bounds__(256) pv_kernel(
    const float* __restrict__ buf_X, const int* __restrict__ buf_y,
    const int* __restrict__ buf_task, const __bf16* __restrict__ w_ops,
    float* __restrict__ out) {
  __shared__ float red[1024];
  const int lane   = threadIdx.x & 31;
  const int wave   = threadIdx.x >> 5;
  const int dcBase = blockIdx.x * 32;
  const int n      = lane & 15;            // B-operand: lane%16 = output col
  const int kAdd   = (lane >> 4) << 4;     // lanes 16-31 carry K+16

  v8f acc00 = {}, acc01 = {}, acc10 = {}, acc11 = {};

  for (int c = wave; c < KCHUNKS; c += 8) {
    const v16bf a0 = *(const v16bf*)(w_ops + ((size_t)(c * 2 + 0) * 32 + lane) * 16);
    const v16bf a1 = *(const v16bf*)(w_ops + ((size_t)(c * 2 + 1) * 32 + lane) * 16);
    const int keyBase = c * 32 + kAdd;
#pragma unroll
    for (int u = 0; u < 2; ++u) {
      const int dcol = dcBase + u * 16 + n;
      v16bf b;
      if (dcol < IMG_D) {
        const float* p = buf_X + (size_t)keyBase * IMG_D + dcol;
#pragma unroll
        for (int j = 0; j < 16; ++j) b[j] = (__bf16)p[(size_t)j * IMG_D];
      } else if (dcol == IMG_D) {
#pragma unroll
        for (int j = 0; j < 16; ++j) b[j] = (__bf16)(float)buf_y[keyBase + j];
      } else if (dcol == IMG_D + 1) {
#pragma unroll
        for (int j = 0; j < 16; ++j) b[j] = (__bf16)(float)buf_task[keyBase + j];
      } else {
#pragma unroll
        for (int j = 0; j < 16; ++j) b[j] = (__bf16)0.0f;
      }
      if (u == 0) { acc00 = wmma_bf16(a0, b, acc00); acc10 = wmma_bf16(a1, b, acc10); }
      else        { acc01 = wmma_bf16(a0, b, acc01); acc11 = wmma_bf16(a1, b, acc11); }
    }
  }

  // cross-wave (split-K) reduction in LDS
  for (int i = threadIdx.x; i < 1024; i += 256) red[i] = 0.0f;
  __syncthreads();
#pragma unroll
  for (int r = 0; r < 8; ++r) {
    atomicAdd(&red[((0 * 2 + 0) * 8 + r) * 32 + lane], acc00[r]);
    atomicAdd(&red[((0 * 2 + 1) * 8 + r) * 32 + lane], acc01[r]);
    atomicAdd(&red[((1 * 2 + 0) * 8 + r) * 32 + lane], acc10[r]);
    atomicAdd(&red[((1 * 2 + 1) * 8 + r) * 32 + lane], acc11[r]);
  }
  __syncthreads();
  for (int i = threadIdx.x; i < 1024; i += 256) {
    const int l  = i & 31;
    const int r  = (i >> 5) & 7;
    const int u  = (i >> 8) & 1;
    const int qt = (i >> 9) & 1;
    const int q    = qt * 16 + r + ((l >> 4) << 3);
    const int dcol = dcBase + u * 16 + (l & 15);
    const float v  = red[i];
    if (dcol < IMG_D)             out[(size_t)q * IMG_D + dcol] = v;
    else if (dcol == IMG_D)       out[NQ * IMG_D + q]      = (float)(int)v;  // y_hop
    else if (dcol == IMG_D + 1)   out[NQ * IMG_D + NQ + q] = (float)(int)v;  // t_hop
  }
}

// -------- kernel 6: fallback select (list buffer) --------
__global__ void __launch_bounds__(256) select_kernel(
    const float* __restrict__ buf_X, const int* __restrict__ buf_y,
    const int* __restrict__ buf_task, const int* __restrict__ li_inds,
    const unsigned char* __restrict__ use_hopfield,
    const float* __restrict__ row_valid, float* __restrict__ out) {
  const int q = blockIdx.x;
  const bool use = (use_hopfield[q] != 0) && (row_valid[q] > 0.5f);
  if (use) return;   // hopfield result already in out
  const int li = li_inds[q];
  for (int d = threadIdx.x; d < IMG_D; d += 256)
    out[(size_t)q * IMG_D + d] = buf_X[(size_t)li * IMG_D + d];
  if (threadIdx.x == 0) {
    out[NQ * IMG_D + q]      = (float)buf_y[li];
    out[NQ * IMG_D + NQ + q] = (float)buf_task[li];
  }
}

extern "C" void kernel_launch(void* const* d_in, const int* in_sizes, int n_in,
                              void* d_out, int out_size, void* d_ws, size_t ws_size,
                              hipStream_t stream) {
  (void)in_sizes; (void)n_in; (void)out_size; (void)ws_size;
  const float* buf_X    = (const float*)d_in[0];
  const int*   buf_y    = (const int*)d_in[1];
  const int*   buf_task = (const int*)d_in[2];
  const float* X        = (const float*)d_in[3];
  const int*   y        = (const int*)d_in[4];
  const int*   task_ids = (const int*)d_in[5];
  const int*   li_inds  = (const int*)d_in[6];
  const unsigned char* use_hopfield = (const unsigned char*)d_in[7]; // jax bool = 1 byte
  float* out = (float*)d_out;

  char* ws = (char*)d_ws;
  float*  scores    = (float*)(ws + WS_SCORES);
  float*  row_max   = (float*)(ws + WS_ROWMAX);
  float*  row_inv   = (float*)(ws + WS_ROWINV);
  float*  row_valid = (float*)(ws + WS_ROWVAL);
  __bf16* q_ops     = (__bf16*)(ws + WS_QOPS);
  __bf16* w_ops     = (__bf16*)(ws + WS_WOPS);

  build_q_ops  <<<(NCHUNK * 2 * 32 * 16 + 255) / 256, 256, 0, stream>>>(X, y, task_ids, q_ops);
  scores_kernel<<<(KEY_TILES + 7) / 8,               256, 0, stream>>>(buf_X, buf_y, buf_task,
                                                                       task_ids, q_ops, scores);
  softmax_stats<<<NQ,                                256, 0, stream>>>(scores, row_max, row_inv,
                                                                       row_valid);
  build_w_ops  <<<(KCHUNKS * 2 * 32 * 16 + 255) / 256, 256, 0, stream>>>(scores, row_max,
                                                                         row_inv, w_ops);
  pv_kernel    <<<NCHUNK,                            256, 0, stream>>>(buf_X, buf_y, buf_task,
                                                                       w_ops, out);
  select_kernel<<<NQ,                                256, 0, stream>>>(buf_X, buf_y, buf_task,
                                                                      li_inds, use_hopfield,
                                                                      row_valid, out);
}
